// MultiHeadAttention_56075093017219
// MI455X (gfx1250) — compile-verified, matrix operand order checked by reading, not run on hardware
//
#include <hip/hip_runtime.h>
#include <stdint.h>

// Problem constants (B=2, S=2048, D=1024, H=16, DH=64)
#define S_LEN 2048
#define D_DIM 1024
#define H_NUM 16
#define DHEAD 64
#define B_NUM 2
#define MROWS (B_NUM * S_LEN)   // 4096

typedef __attribute__((ext_vector_type(16))) __bf16 v16bf;
typedef __attribute__((ext_vector_type(8)))  __bf16 v8bf;
typedef __attribute__((ext_vector_type(8)))  float  v8f;

union FragU { v16bf v; v8bf h[2]; };

// Swap bits 3 and 4 of the k coordinate: after this permutation, lane `half`'s
// 16 fragment elements (k = half*8+{0..7} and 16+half*8+{0..7} per 32-chunk)
// occupy one contiguous 32-byte span. Involution; permutes whole 8-groups.
__device__ __forceinline__ int swz34(int k) {
  return (k & ~24) | ((k & 8) << 1) | ((k & 16) >> 1);
}

__device__ __forceinline__ v8f wmma_bf16(v16bf a, v16bf b, v8f c) {
  return __builtin_amdgcn_wmma_f32_16x16x32_bf16(false, a, false, b, (short)0, c,
                                                 false, false);
}

// Fragment load from swizzled storage: one 32B contiguous span per lane.
__device__ __forceinline__ v16bf load_frag(const __bf16* chunk, int half) {
  const v8bf* p = (const v8bf*)(chunk + half * 16);
  FragU f; f.h[0] = p[0]; f.h[1] = p[1];
  return f.v;
}

// ---- Kernel 0a: W[k][n] fp32 -> Wt[n][swz(k)] bf16 (x3 weights) -------------
__global__ void wt_convert(const float* __restrict__ Wq, const float* __restrict__ Wk,
                           const float* __restrict__ Wv, __bf16* __restrict__ WtBase) {
  const float* W = (blockIdx.z == 0) ? Wq : (blockIdx.z == 1) ? Wk : Wv;
  __bf16* Wt = WtBase + (size_t)blockIdx.z * D_DIM * D_DIM;
  int id = blockIdx.x * blockDim.x + threadIdx.x;   // 131072 per z
  int n = id >> 7, k8 = (id & 127) * 8;
  v8f f;
#pragma unroll
  for (int j = 0; j < 8; ++j) f[j] = W[(size_t)(k8 + j) * D_DIM + n];
  *(v8bf*)(Wt + (size_t)n * D_DIM + swz34(k8)) = __builtin_convertvector(f, v8bf);
}

// ---- Kernel 0b: X[m][k] fp32 -> Xb[m][swz(k)] bf16 (q,k,v activations) ------
__global__ void x_convert(const float* __restrict__ Xq, const float* __restrict__ Xk,
                          const float* __restrict__ Xv, __bf16* __restrict__ XbBase) {
  const float* X = (blockIdx.z == 0) ? Xq : (blockIdx.z == 1) ? Xk : Xv;
  __bf16* Xb = XbBase + (size_t)blockIdx.z * MROWS * D_DIM;
  int id = blockIdx.x * blockDim.x + threadIdx.x;   // 524288 per z
  int m = id >> 7, k8 = (id & 127) * 8;
  v8f f = *(const v8f*)(X + (size_t)m * D_DIM + k8);
  *(v8bf*)(Xb + (size_t)m * D_DIM + swz34(k8)) = __builtin_convertvector(f, v8bf);
}

// ---- Kernel 1: QKV projection, 32x64 tile per wave --------------------------
// z=0: Q*0.125 -> Qs[head][s][swz(dh)]
// z=1: K       -> Ks[head][s][swz(dh)]
// z=2: V       -> Vt[head][dh][swz(s)]
__global__ void qkv_proj(const __bf16* __restrict__ XbBase,
                         const float* __restrict__ bq, const float* __restrict__ bk,
                         const float* __restrict__ bv,
                         const __bf16* __restrict__ WtBase,
                         __bf16* __restrict__ Qs, __bf16* __restrict__ Ks,
                         __bf16* __restrict__ Vt) {
  const int z = blockIdx.z;
  const float*  bia = (z == 0) ? bq : (z == 1) ? bk : bv;
  const __bf16* Xb  = XbBase + (size_t)z * MROWS * D_DIM;
  const __bf16* Wt  = WtBase + (size_t)z * D_DIM * D_DIM;
  const int m0 = blockIdx.x * 32, n0 = blockIdx.y * 64;
  const int lane = threadIdx.x & 31, half = lane >> 4, lm = lane & 15;

  const __bf16* arow0 = Xb + (size_t)(m0 + lm) * D_DIM;
  const __bf16* arow1 = arow0 + (size_t)16 * D_DIM;
  const __bf16* brow[4];
#pragma unroll
  for (int ni = 0; ni < 4; ++ni) brow[ni] = Wt + (size_t)(n0 + ni * 16 + lm) * D_DIM;

  v8f cc[2][4] = {};
  for (int k0 = 0; k0 < D_DIM; k0 += 32) {
    v16bf a0 = load_frag(arow0 + k0, half);
    v16bf a1 = load_frag(arow1 + k0, half);
#pragma unroll
    for (int ni = 0; ni < 4; ++ni) {
      v16bf b = load_frag(brow[ni] + k0, half);
      cc[0][ni] = wmma_bf16(a0, b, cc[0][ni]);
      cc[1][ni] = wmma_bf16(a1, b, cc[1][ni]);
    }
  }

  // Epilogue: uniform-z paths (no per-element branches).
  if (z == 2) {
#pragma unroll
    for (int ni = 0; ni < 4; ++ni) {
      const int ncol = n0 + ni * 16 + lm;
      const int h = ncol >> 6, dh = ncol & (DHEAD - 1);
      const float bias = bia[ncol];
#pragma unroll
      for (int mi = 0; mi < 2; ++mi)
#pragma unroll
        for (int r = 0; r < 8; ++r) {
          int    mg = m0 + mi * 16 + r + 8 * half;
          int    bb = mg >> 11, s = mg & (S_LEN - 1);
          size_t head = (size_t)(bb * H_NUM + h);
          Vt[(head * DHEAD + dh) * S_LEN + swz34(s)] = (__bf16)(cc[mi][ni][r] + bias);
        }
    }
  } else {
    __bf16* dst = (z == 0) ? Qs : Ks;
    const float mul = (z == 0) ? 0.125f : 1.0f;   // fold 1/sqrt(DH) into Q
#pragma unroll
    for (int ni = 0; ni < 4; ++ni) {
      const int ncol = n0 + ni * 16 + lm;
      const int h = ncol >> 6, dhs = swz34(ncol & (DHEAD - 1));
      const float bias = bia[ncol];
#pragma unroll
      for (int mi = 0; mi < 2; ++mi)
#pragma unroll
        for (int r = 0; r < 8; ++r) {
          int    mg = m0 + mi * 16 + r + 8 * half;
          int    bb = mg >> 11, s = mg & (S_LEN - 1);
          size_t head = (size_t)(bb * H_NUM + h);
          dst[(head * S_LEN + s) * DHEAD + dhs] = (__bf16)((cc[mi][ni][r] + bias) * mul);
        }
    }
  }
}

// ---- Kernel 2: flash attention, 32 queries per wave -------------------------
// S^T = K x Q^T (all fragments contiguous swizzled loads); online softmax with
// per-q state lane-resident; P's B-fragment built lane-locally from the S^T
// f32 accumulators (C layout == B-fragment layout, no LDS / no shuffles);
// O^T += Vt x P. Each K/V fragment feeds both query groups (2x reuse).
__global__ void flash_attn(const __bf16* __restrict__ Qs, const __bf16* __restrict__ Ks,
                           const __bf16* __restrict__ Vt,
                           const float* __restrict__ resid, float* __restrict__ out) {
  const int qt = blockIdx.x, h = blockIdx.y, b = blockIdx.z;
  const int lane = threadIdx.x & 31, half = lane >> 4, lm = lane & 15;
  const size_t head = (size_t)(b * H_NUM + h);
  const __bf16* Qh = Qs + head * (size_t)S_LEN * DHEAD;
  const __bf16* Kh = Ks + head * (size_t)S_LEN * DHEAD;
  const __bf16* Vh = Vt + head * (size_t)DHEAD * S_LEN;
  const int q0 = qt * 32;

  v16bf qb[2][2];
#pragma unroll
  for (int g = 0; g < 2; ++g) {
    const __bf16* qrow = Qh + (size_t)(q0 + g * 16 + lm) * DHEAD;
    qb[g][0] = load_frag(qrow, half);        // dh 0..31
    qb[g][1] = load_frag(qrow + 32, half);   // dh 32..63
  }

  v8f   ot[2][4] = {};
  float mrow[2] = {-3.0e38f, -3.0e38f};
  float lrow[2] = {0.0f, 0.0f};

  for (int kc = 0; kc < S_LEN; kc += 64) {
    // --- scores^T: 4 key tiles x 2 query groups (K fragments reused) ---
    v8f st[2][4];
#pragma unroll
    for (int t = 0; t < 4; ++t) {
      const __bf16* krow = Kh + (size_t)(kc + t * 16 + lm) * DHEAD;
      v16bf ka0 = load_frag(krow, half);
      v16bf ka1 = load_frag(krow + 32, half);
#pragma unroll
      for (int g = 0; g < 2; ++g) {
        v8f c = {};
        c = wmma_bf16(ka0, qb[g][0], c);
        c = wmma_bf16(ka1, qb[g][1], c);
        st[g][t] = c;   // lane: col q = q0+g*16+lm; VGPR r: key = kc+t*16+r+8*half
      }
    }

    // --- online softmax per query group ---
    v16bf pb[2][2];
#pragma unroll
    for (int g = 0; g < 2; ++g) {
      float mx = mrow[g];
#pragma unroll
      for (int t = 0; t < 4; ++t)
#pragma unroll
        for (int r = 0; r < 8; ++r) mx = fmaxf(mx, st[g][t][r]);
      mx = fmaxf(mx, __shfl_xor(mx, 16, 32));
      const float corr = __expf(mrow[g] - mx);
      mrow[g] = mx;
      float rs = 0.0f;
#pragma unroll
      for (int t = 0; t < 4; ++t)
#pragma unroll
        for (int r = 0; r < 8; ++r) {
          float p = __expf(st[g][t][r] - mx);
          st[g][t][r] = p;
          rs += p;
        }
      rs += __shfl_xor(rs, 16, 32);
      lrow[g] = lrow[g] * corr + rs;
#pragma unroll
      for (int t = 0; t < 4; ++t) ot[g][t] = ot[g][t] * corr;
#pragma unroll
      for (int kk = 0; kk < 2; ++kk) {
        FragU f;
        f.h[0] = __builtin_convertvector(st[g][2 * kk], v8bf);
        f.h[1] = __builtin_convertvector(st[g][2 * kk + 1], v8bf);
        pb[g][kk] = f.v;
      }
    }

    // --- O^T += Vt x P (V fragments reused across query groups) ---
#pragma unroll
    for (int t = 0; t < 4; ++t) {
      const __bf16* vrow = Vh + (size_t)(t * 16 + lm) * S_LEN + kc;
      v16bf va0 = load_frag(vrow, half);
      v16bf va1 = load_frag(vrow + 32, half);
#pragma unroll
      for (int g = 0; g < 2; ++g) {
        ot[g][t] = wmma_bf16(va0, pb[g][0], ot[g][t]);
        ot[g][t] = wmma_bf16(va1, pb[g][1], ot[g][t]);
      }
    }
  }

  // --- epilogue: normalize, add residual, store fp32 (contiguous in dh) ---
#pragma unroll
  for (int g = 0; g < 2; ++g) {
    const float  linv = 1.0f / lrow[g];
    const int    qg = q0 + g * 16 + lm;
    const size_t rowbase = ((size_t)b * S_LEN + qg) * D_DIM + h * DHEAD;
#pragma unroll
    for (int t = 0; t < 4; ++t)
#pragma unroll
      for (int r = 0; r < 8; ++r) {
        int dh = t * 16 + 8 * half + r;
        out[rowbase + dh] = ot[g][t][r] * linv + resid[rowbase + dh];
      }
  }
}

// -----------------------------------------------------------------------------
extern "C" void kernel_launch(void* const* d_in, const int* in_sizes, int n_in,
                              void* d_out, int out_size, void* d_ws, size_t ws_size,
                              hipStream_t stream) {
  (void)in_sizes; (void)n_in; (void)out_size; (void)ws_size;
  const float* q  = (const float*)d_in[0];
  const float* k  = (const float*)d_in[1];
  const float* v  = (const float*)d_in[2];
  const float* Wq = (const float*)d_in[3];
  const float* bq = (const float*)d_in[4];
  const float* Wk = (const float*)d_in[5];
  const float* bk = (const float*)d_in[6];
  const float* Wv = (const float*)d_in[7];
  const float* bv = (const float*)d_in[8];
  float* out = (float*)d_out;

  __bf16* ws = (__bf16*)d_ws;
  const size_t DD  = (size_t)D_DIM * D_DIM;                  // 1 Mi
  const size_t MD  = (size_t)MROWS * D_DIM;                  // 4 Mi
  const size_t QKV = (size_t)B_NUM * H_NUM * S_LEN * DHEAD;  // 4 Mi
  __bf16* WtBase = ws;             // 3*DD
  __bf16* XbBase = WtBase + 3 * DD;
  __bf16* QsB = XbBase + 3 * MD;
  __bf16* KsB = QsB + QKV;
  __bf16* VtB = KsB + QKV;

  wt_convert<<<dim3(512, 1, 3), 256, 0, stream>>>(Wq, Wk, Wv, WtBase);
  x_convert<<<dim3(2048, 1, 3), 256, 0, stream>>>(q, k, v, XbBase);
  qkv_proj<<<dim3(MROWS / 32, D_DIM / 64, 3), 32, 0, stream>>>(
      XbBase, bq, bk, bv, WtBase, QsB, KsB, VtB);
  flash_attn<<<dim3(S_LEN / 32, H_NUM, B_NUM), 32, 0, stream>>>(QsB, KsB, VtB, q, out);
}
